// RGCNBlock_87282325389872
// MI455X (gfx1250) — matrix operand order repaired and checked by reading, not compiled, not run on hardware
//
#include <hip/hip_runtime.h>
#include <hip/hip_bf16.h>

#define NNODES 50000
#define FEAT   128
#define KNBR   16
#define KW     (KNBR * FEAT)   // 2048
#define NB_RED 200
#define BN_EPS 1e-5f
#define LDS_STRIDE 272         // 128 bf16 (256B) + 16B pad -> conflict-free b128 LDS reads

typedef __attribute__((ext_vector_type(16))) __bf16 bf16x16;
typedef __attribute__((ext_vector_type(8)))  __bf16 bf16x8;
typedef __attribute__((ext_vector_type(8)))  float  f32x8;

#define SHUF16(lo, hi) __builtin_shufflevector(lo, hi, 0,1,2,3,4,5,6,7,8,9,10,11,12,13,14,15)

// ---------------- BN statistics: deterministic two-pass reduction ----------------

__global__ void reduce_stats(const float* __restrict__ in, float* __restrict__ partials, int n) {
    int f = threadIdx.x;                       // 128 threads = 128 features
    float s = 0.f, ss = 0.f;
    for (int r = blockIdx.x; r < n; r += gridDim.x) {
        float v = in[(size_t)r * FEAT + f];
        s += v; ss += v * v;
    }
    partials[blockIdx.x * (2 * FEAT) + f]        = s;
    partials[blockIdx.x * (2 * FEAT) + FEAT + f] = ss;
}

__global__ void finalize_stats(const float* __restrict__ partials,
                               const float* __restrict__ gamma,
                               const float* __restrict__ beta,
                               float* __restrict__ ac, int nblocks, float invn) {
    int f = threadIdx.x;
    float s = 0.f, ss = 0.f;
    for (int b = 0; b < nblocks; ++b) {        // fixed order -> deterministic
        s  += partials[b * (2 * FEAT) + f];
        ss += partials[b * (2 * FEAT) + FEAT + f];
    }
    float mean = s * invn;
    float var  = ss * invn - mean * mean;
    float a    = gamma[f] * rsqrtf(var + BN_EPS);
    ac[f]        = a;                          // scale
    ac[FEAT + f] = beta[f] - mean * a;         // shift
}

// ---------------- fused BN + exact GELU + f32->bf16, with zero pad row ----------------

__global__ void bn_gelu_tobf16(const float* __restrict__ in, const float* __restrict__ ac,
                               __bf16* __restrict__ h, int n) {
    size_t total = (size_t)(n + 1) * FEAT;     // +1 zero pad row (h_pad semantics)
    size_t lim   = (size_t)n * FEAT;
    for (size_t i = (size_t)blockIdx.x * blockDim.x + threadIdx.x; i < total;
         i += (size_t)gridDim.x * blockDim.x) {
        float g = 0.f;
        if (i < lim) {
            int f   = (int)(i & (FEAT - 1));
            float y = in[i] * ac[f] + ac[FEAT + f];
            g = 0.5f * y * (1.0f + erff(y * 0.70710678118654752f));  // exact GELU
        }
        h[i] = (__bf16)g;
    }
}

__global__ void conv_w_bf16(const float* __restrict__ w, __bf16* __restrict__ wb, int n) {
    for (int i = blockIdx.x * blockDim.x + threadIdx.x; i < n; i += gridDim.x * blockDim.x)
        wb[i] = (__bf16)w[i];
}

// ---------------- async global->LDS staging (CDNA5, ASYNCcnt-tracked) ----------------
// LDS aperture: generic addr[31:0] == LDS byte offset (ISA 10.2), so truncating the
// generic pointer gives the 32-bit LDS address the instruction needs.

__device__ __forceinline__ void async_copy16(const void* g, void* l) {
    unsigned lds = (unsigned)(size_t)l;
    asm volatile("global_load_async_to_lds_b128 %0, %1, off"
                 :: "v"(lds), "v"(g) : "memory");
}

__device__ __forceinline__ void wait_async0() {
    asm volatile("s_wait_asynccnt 0x0" ::: "memory");
}

// ---------------- gathered GEMM via V_WMMA_F32_16X16X32_BF16 ----------------
// Workgroup = 8 waves; each wave owns TWO 16-node tiles (M=32, 128 acc VGPRs).
// W slice for neighbor slot j (128x128 bf16 = 32KB) is double-buffered in LDS and
// staged with global_load_async_to_lds_b128 (no VGPR round-trip); prefetch for
// slice j+1 is issued at the top of iter j and waited (ASYNCcnt) only at the
// barrier after 64 WMMAs. Each LDS B fragment feeds two WMMAs.

__global__ __launch_bounds__(256)
void rgcn_gemm(const __bf16* __restrict__ h,     // (NNODES+1) x 128 bf16
               const __bf16* __restrict__ Wb,    // 128 x 2048 bf16, row-major (o, k)
               const int*    __restrict__ graph, // NNODES x 16 x 2 int32
               const float*  __restrict__ bias,  // 128
               const float*  __restrict__ resid, // NNODES x 128 or nullptr
               float*        __restrict__ out,   // NNODES x 128
               int ntiles) {
    __shared__ __align__(16) unsigned char smem[2][128 * LDS_STRIDE];  // 68 KB

    const int tid  = threadIdx.x;
    const int lane = tid & 31;
    const int m    = lane & 15;
    const int half = lane >> 4;

    const int pair = blockIdx.x * 8 + (tid >> 5);
    int t0 = pair * 2, t1 = pair * 2 + 1;
    const bool act0 = t0 < ntiles;
    const bool act1 = t1 < ntiles;
    if (!act0) t0 = ntiles - 1;                // clamp: keep addresses valid,
    if (!act1) t1 = ntiles - 1;                // all waves stay in barrier lockstep

    const int* gp0 = graph + (size_t)(t0 * 16 + m) * (2 * KNBR) + 1;  // graph[...,1]
    const int* gp1 = graph + (size_t)(t1 * 16 + m) * (2 * KNBR) + 1;

    // W staging role: thread covers half a slice row (64 bf16 = 128B contiguous)
    const int wrow = tid >> 1;
    const int wsub = (tid & 1);
    const __bf16*  wsrc0 = Wb + (size_t)wrow * KW + wsub * 64;
    unsigned char* ldst0 = &smem[0][wrow * LDS_STRIDE + wsub * 128];
    unsigned char* ldst1 = &smem[1][wrow * LDS_STRIDE + wsub * 128];

    f32x8 acc0[8], acc1[8];
    #pragma unroll
    for (int t = 0; t < 8; ++t) {
        acc0[t] = (f32x8){0.f,0.f,0.f,0.f,0.f,0.f,0.f,0.f};
        acc1[t] = (f32x8){0.f,0.f,0.f,0.f,0.f,0.f,0.f,0.f};
    }

    // prologue: async-stage W slice 0 into smem[0]
    #pragma unroll
    for (int i = 0; i < 8; ++i) async_copy16(wsrc0 + i * 8, ldst0 + i * 16);
    int idx0 = gp0[0];
    int idx1 = gp1[0];
    wait_async0();
    __syncthreads();

    for (int j = 0; j < KNBR; ++j) {
        // issue async prefetch of next W slice into the other LDS buffer
        if (j < KNBR - 1) {
            const __bf16*  wsrc = wsrc0 + (j + 1) * FEAT;
            unsigned char* dst  = ((j + 1) & 1) ? ldst1 : ldst0;
            #pragma unroll
            for (int i = 0; i < 8; ++i) async_copy16(wsrc + i * 8, dst + i * 16);
        }
        const __bf16* ar0 = h + (size_t)idx0 * FEAT;   // gathered neighbor rows
        const __bf16* ar1 = h + (size_t)idx1 * FEAT;
        if (j < KNBR - 1) { idx0 = gp0[2 * (j + 1)]; idx1 = gp1[2 * (j + 1)]; }  // rotate

        const unsigned char* wlds = &smem[j & 1][0];
        #pragma unroll
        for (int f4 = 0; f4 < 4; ++f4) {
            // A fragments (16x32 bf16): lane holds K [base,base+8) and [base+16,base+24)
            bf16x8 a0l = *(const bf16x8*)(ar0 + f4 * 32 + half * 8);
            bf16x8 a0h = *(const bf16x8*)(ar0 + f4 * 32 + 16 + half * 8);
            bf16x8 a1l = *(const bf16x8*)(ar1 + f4 * 32 + half * 8);
            bf16x8 a1h = *(const bf16x8*)(ar1 + f4 * 32 + 16 + half * 8);
            bf16x16 A0 = SHUF16(a0l, a0h);
            bf16x16 A1 = SHUF16(a1l, a1h);
            #pragma unroll
            for (int t = 0; t < 8; ++t) {
                // B fragment (32x16): lane holds K [base,base+8) and [base+16,base+24) of W row o=t*16+m
                const unsigned char* p =
                    wlds + (size_t)(t * 16 + m) * LDS_STRIDE + f4 * 64 + half * 16;
                bf16x8 bl = *(const bf16x8*)(p);
                bf16x8 bh = *(const bf16x8*)(p + 32);
                bf16x16 B = SHUF16(bl, bh);
                acc0[t] = __builtin_amdgcn_wmma_f32_16x16x32_bf16(
                              false, A0, false, B, (short)0, acc0[t], false, false);
                acc1[t] = __builtin_amdgcn_wmma_f32_16x16x32_bf16(
                              false, A1, false, B, (short)0, acc1[t], false, false);
            }
        }
        wait_async0();     // staged slice j+1 fully in LDS
        __syncthreads();   // double buffer: one barrier per iteration is sufficient
    }

    // C/D layout: VGPR r, lane -> M = r + half*8, Ncol = m
    #pragma unroll
    for (int t = 0; t < 8; ++t) {
        int col  = t * 16 + m;
        float bv = bias[col];
        #pragma unroll
        for (int r = 0; r < 8; ++r) {
            int rr = r + half * 8;
            if (act0) {
                size_t o = (size_t)(t0 * 16 + rr) * FEAT + col;
                float v  = acc0[t][r] + bv;
                if (resid) v += resid[o];
                out[o] = v;
            }
            if (act1) {
                size_t o = (size_t)(t1 * 16 + rr) * FEAT + col;
                float v  = acc1[t][r] + bv;
                if (resid) v += resid[o];
                out[o] = v;
            }
        }
    }
}

// ---------------- host launch ----------------

extern "C" void kernel_launch(void* const* d_in, const int* in_sizes, int n_in,
                              void* d_out, int out_size, void* d_ws, size_t ws_size,
                              hipStream_t stream) {
    const float* x      = (const float*)d_in[0];
    const int*   graph  = (const int*)  d_in[1];
    const float* gamma1 = (const float*)d_in[2];
    const float* beta1  = (const float*)d_in[3];
    const float* W1     = (const float*)d_in[4];
    const float* b1     = (const float*)d_in[5];
    const float* gamma2 = (const float*)d_in[6];
    const float* beta2  = (const float*)d_in[7];
    const float* W2     = (const float*)d_in[8];
    const float* b2     = (const float*)d_in[9];
    float* out = (float*)d_out;

    // workspace carving (all offsets 256B aligned)
    char* ws = (char*)d_ws;
    size_t off = 0;
    __bf16* w1b = (__bf16*)(ws + off); off += (size_t)FEAT * KW * 2;          // 512 KB
    __bf16* w2b = (__bf16*)(ws + off); off += (size_t)FEAT * KW * 2;          // 512 KB
    float* ac1  = (float*)(ws + off);  off += 2 * FEAT * sizeof(float);
    float* ac2  = (float*)(ws + off);  off += 2 * FEAT * sizeof(float);
    off = (off + 255) & ~(size_t)255;
    float* partials = (float*)(ws + off); off += (size_t)NB_RED * 2 * FEAT * 4; // 200 KB
    off = (off + 255) & ~(size_t)255;
    __bf16* hbuf = (__bf16*)(ws + off); off += (size_t)(NNODES + 1) * FEAT * 2; // 12.8 MB
    off = (off + 255) & ~(size_t)255;
    float* tbuf  = (float*)(ws + off);  off += (size_t)NNODES * FEAT * 4;       // 25.6 MB

    const int ntiles = NNODES / 16;                    // 3125
    const int npairs = (ntiles + 1) / 2;               // 1563 (2 tiles per wave)
    const int gblk   = (npairs + 7) / 8;               // 8 waves per block
    const float invn = 1.0f / (float)NNODES;

    conv_w_bf16<<<256, 256, 0, stream>>>(W1, w1b, FEAT * KW);
    conv_w_bf16<<<256, 256, 0, stream>>>(W2, w2b, FEAT * KW);

    // layer 1
    reduce_stats  <<<NB_RED, FEAT, 0, stream>>>(x, partials, NNODES);
    finalize_stats<<<1,      FEAT, 0, stream>>>(partials, gamma1, beta1, ac1, NB_RED, invn);
    bn_gelu_tobf16<<<1024,   256,  0, stream>>>(x, ac1, hbuf, NNODES);
    rgcn_gemm     <<<gblk,   256,  0, stream>>>(hbuf, w1b, graph, b1, nullptr, tbuf, ntiles);

    // layer 2 + residual
    reduce_stats  <<<NB_RED, FEAT, 0, stream>>>(tbuf, partials, NNODES);
    finalize_stats<<<1,      FEAT, 0, stream>>>(partials, gamma2, beta2, ac2, NB_RED, invn);
    bn_gelu_tobf16<<<1024,   256,  0, stream>>>(tbuf, ac2, hbuf, NNODES);
    rgcn_gemm     <<<gblk,   256,  0, stream>>>(hbuf, w2b, graph, b2, x, out, ntiles);
}
